// MultiHeadAttention_29927332118923
// MI455X (gfx1250) — compile-verified, matrix operand order checked by reading, not run on hardware
//
#include <hip/hip_runtime.h>
#include <math.h>

#define B_  2
#define L_  2048
#define E_  256
#define H_  8
#define DS_ 256
#define EH_ (E_ * H_)   // 2048
#define BL_ (B_ * L_)   // 4096

typedef __bf16 bf16_t;
typedef __attribute__((ext_vector_type(16))) __bf16 v16bf;
typedef __attribute__((ext_vector_type(8)))  __bf16 v8bf;
typedef __attribute__((ext_vector_type(4)))  __bf16 v4bf;
typedef __attribute__((ext_vector_type(8)))  float  v8f;
typedef __attribute__((ext_vector_type(4)))  float  v4f;

union Frag16 { v16bf v; v8bf h[2]; };

// Load a 16-element bf16 WMMA fragment for one lane.
// Per CDNA5 ISA layout: elements 0..7 = K base+0..7, elements 8..15 = K base+16..23.
// Caller passes p = row_base + (lane>>4)*8 so the two halves land at the right K.
static __device__ inline v16bf load_frag(const bf16_t* p) {
  Frag16 f;
  f.h[0] = *reinterpret_cast<const v8bf*>(p);
  f.h[1] = *reinterpret_cast<const v8bf*>(p + 16);
  return f.v;
}

static __device__ inline v8f wmma_bf16(v16bf a, v16bf b, v8f c) {
  return __builtin_amdgcn_wmma_f32_16x16x32_bf16(false, a, false, b, (short)0, c,
                                                 false, false);
}

// ---------------- fp32 -> bf16 convert ----------------
__global__ __launch_bounds__(256) void cvt_bf16_kernel(const float* __restrict__ in,
                                                       bf16_t* __restrict__ out, int n) {
  int i = (blockIdx.x * blockDim.x + threadIdx.x) * 4;
  if (i < n) {
    v4f x = *reinterpret_cast<const v4f*>(in + i);
    v4bf y;
    y[0] = (bf16_t)x[0]; y[1] = (bf16_t)x[1];
    y[2] = (bf16_t)x[2]; y[3] = (bf16_t)x[3];
    *reinterpret_cast<v4bf*>(out + i) = y;
  }
}

// ---------------- tiled transpose fp32 (K x N) -> bf16 (N x K) ----------------
// grid = (N/32, K/32), block = 256. Coalesced on both sides via 32x33 LDS tile.
__global__ __launch_bounds__(256) void transpose_bf16_kernel(const float* __restrict__ in,
                                                             bf16_t* __restrict__ outT,
                                                             int K, int N) {
  __shared__ bf16_t tile[32][33];
  const int n0 = blockIdx.x * 32;
  const int k0 = blockIdx.y * 32;
  const int tx = threadIdx.x & 31;
  const int ty = threadIdx.x >> 5;   // 0..7
#pragma unroll
  for (int r = 0; r < 32; r += 8)
    tile[ty + r][tx] = (bf16_t)in[(size_t)(k0 + ty + r) * N + n0 + tx];
  __syncthreads();
#pragma unroll
  for (int r = 0; r < 32; r += 8)
    outT[(size_t)(n0 + ty + r) * K + k0 + tx] = tile[tx][ty + r];
}

// ---------------- QKV projection: Y = Xb @ W + b, head-split store ----------------
// grid = (BL_/16, EH_/128), block = 256 (8 waves, one 16x16 N-tile each)
__global__ __launch_bounds__(256) void qkv_proj_kernel(const bf16_t* __restrict__ xb,
                                                       const bf16_t* __restrict__ wT,
                                                       const float* __restrict__ bias,
                                                       bf16_t* __restrict__ out,
                                                       int vtrans) {
  const int tid  = threadIdx.x;
  const int lane = tid & 31;
  const int wave = tid >> 5;
  const int mrow = lane & 15;
  const int hh   = lane >> 4;
  const int nn   = lane & 15;
  const int row0 = blockIdx.x * 16;
  const int col0 = blockIdx.y * 128 + wave * 16;

  v8f acc = {};
#pragma unroll
  for (int c = 0; c < 8; ++c) {
    v16bf a = load_frag(xb + (size_t)(row0 + mrow) * E_ + c * 32 + hh * 8);
    v16bf b = load_frag(wT + (size_t)(col0 + nn) * E_ + c * 32 + hh * 8);
    acc = wmma_bf16(a, b, acc);
  }
  const float bv  = bias[col0 + nn];
  const int colg  = col0 + nn;
  const int hIdx  = colg >> 8;    // head
  const int e     = colg & 255;   // within-head dim
#pragma unroll
  for (int r = 0; r < 8; ++r) {
    int grow = row0 + hh * 8 + r;
    int b_   = grow >> 11;        // batch
    int lrow = grow & 2047;       // sequence pos
    float val = acc[r] + bv;
    if (vtrans)  // v stored (B,H,E,L) so attention B-fragments are K-contiguous
      out[((size_t)(b_ * H_ + hIdx) * E_ + e) * L_ + lrow] = (bf16_t)val;
    else         // q/k stored (B,H,L,E)
      out[((size_t)(b_ * H_ + hIdx) * L_ + lrow) * E_ + e] = (bf16_t)val;
  }
}

// ---------------- fused attention: scores -> softmax -> attn out + context ----------------
#define SC_BYTES (16 * L_ * 4)            // 131072: fp32 scores, 16 rows x 2048
#define PR_BYTES (16 * L_ * 2)            // 65536:  bf16 probs
#define RED_OFF  (SC_BYTES + PR_BYTES)
#define SMEM_ATTN (RED_OFF + 512 * 4)     // + reduction scratch

__global__ __launch_bounds__(256) void attn_kernel(const bf16_t* __restrict__ qs,
                                                   const bf16_t* __restrict__ ks,
                                                   const bf16_t* __restrict__ vt,
                                                   const unsigned char* __restrict__ mask,
                                                   float* __restrict__ attn_out,
                                                   bf16_t* __restrict__ ctx) {
  extern __shared__ char smem[];
  float*  scores = (float*)smem;
  bf16_t* probs  = (bf16_t*)(smem + SC_BYTES);
  float*  red    = (float*)(smem + RED_OFF);

  const int tid  = threadIdx.x;
  const int lane = tid & 31;
  const int wave = tid >> 5;
  const int mrow = lane & 15;
  const int hh   = lane >> 4;
  const int nn   = lane & 15;
  const int q0   = blockIdx.x * 16;
  const int bh   = blockIdx.y;        // b*H + h
  const int b_   = bh >> 3;
  const int h_   = bh & 7;

  // Q fragments for this 16-row tile stay resident in VGPRs (8 x v16bf = 64 VGPRs)
  v16bf aQ[8];
#pragma unroll
  for (int c = 0; c < 8; ++c)
    aQ[c] = load_frag(qs + ((size_t)bh * L_ + q0 + mrow) * E_ + c * 32 + hh * 8);

  // ---- Phase 1: S = (Q Kt)/16, masked, into LDS ----
  for (int j = wave; j < L_ / 16; j += 8) {
    v8f acc = {};
#pragma unroll
    for (int c = 0; c < 8; ++c) {
      v16bf bK = load_frag(ks + ((size_t)bh * L_ + j * 16 + nn) * E_ + c * 32 + hh * 8);
      acc = wmma_bf16(aQ[c], bK, acc);
    }
#pragma unroll
    for (int r = 0; r < 8; ++r) {
      int m   = hh * 8 + r;
      int key = j * 16 + nn;
      // mask is streamed once: non-temporal load
      unsigned char mk =
          __builtin_nontemporal_load(mask + ((size_t)b_ * L_ + q0 + m) * L_ + key);
      scores[m * L_ + key] = mk ? -1e9f : acc[r] * 0.0625f;  // 1/sqrt(256)
    }
  }
  __syncthreads();

  // ---- Phase 2: row softmax; write fp32 attn to global (NT), bf16 probs to LDS ----
  {
    const int row = tid >> 4;
    const int l16 = tid & 15;
    float* srow = scores + row * L_ + l16 * 128;
    float mx = -3.4e38f;
#pragma unroll 8
    for (int i = 0; i < 128; i += 4) {
      v4f x = *reinterpret_cast<const v4f*>(srow + i);
      mx = fmaxf(mx, fmaxf(fmaxf(x[0], x[1]), fmaxf(x[2], x[3])));
    }
    red[tid] = mx;
    __syncthreads();
    float rowmax = -3.4e38f;
#pragma unroll
    for (int i = 0; i < 16; ++i) rowmax = fmaxf(rowmax, red[row * 16 + i]);

    float sum = 0.f;
#pragma unroll 4
    for (int i = 0; i < 128; i += 4) {
      v4f x = *reinterpret_cast<const v4f*>(srow + i);
      v4f e4;
      e4[0] = __expf(x[0] - rowmax); e4[1] = __expf(x[1] - rowmax);
      e4[2] = __expf(x[2] - rowmax); e4[3] = __expf(x[3] - rowmax);
      sum += e4[0] + e4[1] + e4[2] + e4[3];
      *reinterpret_cast<v4f*>(srow + i) = e4;
    }
    red[256 + tid] = sum;
    __syncthreads();
    float tot = 0.f;
#pragma unroll
    for (int i = 0; i < 16; ++i) tot += red[256 + row * 16 + i];
    float inv = 1.f / tot;

    float*  arow = attn_out + ((size_t)bh * L_ + q0 + row) * L_ + l16 * 128;
    bf16_t* prow = probs + row * L_ + l16 * 128;
#pragma unroll 4
    for (int i = 0; i < 128; i += 4) {
      v4f x = *reinterpret_cast<const v4f*>(srow + i);
      v4f p;
      p[0] = x[0] * inv; p[1] = x[1] * inv; p[2] = x[2] * inv; p[3] = x[3] * inv;
      // attn (268MB) is write-once and larger than L2: stream it non-temporally
      __builtin_nontemporal_store(p, reinterpret_cast<v4f*>(arow + i));
      v4bf pb;
      pb[0] = (bf16_t)p[0]; pb[1] = (bf16_t)p[1];
      pb[2] = (bf16_t)p[2]; pb[3] = (bf16_t)p[3];
      *reinterpret_cast<v4bf*>(prow + i) = pb;
    }
  }
  __syncthreads();

  // ---- Phase 3: C = P @ V, P from LDS, V^T fragments from global ----
  v8f accC0 = {}, accC1 = {};
  const int nt0 = wave * 2;
#pragma unroll 4
  for (int c = 0; c < 64; ++c) {
    v16bf aP  = load_frag(probs + mrow * L_ + c * 32 + hh * 8);
    v16bf bV0 = load_frag(vt + ((size_t)bh * E_ + nt0 * 16 + nn) * L_ + c * 32 + hh * 8);
    accC0 = wmma_bf16(aP, bV0, accC0);
    v16bf bV1 = load_frag(vt + ((size_t)bh * E_ + (nt0 + 1) * 16 + nn) * L_ + c * 32 + hh * 8);
    accC1 = wmma_bf16(aP, bV1, accC1);
  }
#pragma unroll
  for (int r = 0; r < 8; ++r) {
    int m = hh * 8 + r;
    size_t base = (size_t)(b_ * L_ + q0 + m) * EH_ + h_ * E_;
    ctx[base + nt0 * 16 + nn]       = (bf16_t)accC0[r];
    ctx[base + (nt0 + 1) * 16 + nn] = (bf16_t)accC1[r];
  }
}

// ---------------- output projection + bias + residual + LayerNorm ----------------
// grid = BL_/16, block = 256
__global__ __launch_bounds__(256) void out_proj_ln_kernel(const bf16_t* __restrict__ ctx,
                                                          const bf16_t* __restrict__ woT,
                                                          const float* __restrict__ bo,
                                                          const float* __restrict__ resq,
                                                          const float* __restrict__ gamma,
                                                          const float* __restrict__ beta,
                                                          float* __restrict__ out) {
  __shared__ float xs[16 * DS_];
  __shared__ float red[512];
  const int tid  = threadIdx.x;
  const int lane = tid & 31;
  const int wave = tid >> 5;
  const int mrow = lane & 15;
  const int hh   = lane >> 4;
  const int nn   = lane & 15;
  const int row0 = blockIdx.x * 16;
  const int nt0  = wave * 2;

  v8f acc0 = {}, acc1 = {};
#pragma unroll 4
  for (int c = 0; c < 64; ++c) {
    v16bf a  = load_frag(ctx + (size_t)(row0 + mrow) * EH_ + c * 32 + hh * 8);
    v16bf b0 = load_frag(woT + (size_t)(nt0 * 16 + nn) * EH_ + c * 32 + hh * 8);
    acc0 = wmma_bf16(a, b0, acc0);
    v16bf b1 = load_frag(woT + (size_t)((nt0 + 1) * 16 + nn) * EH_ + c * 32 + hh * 8);
    acc1 = wmma_bf16(a, b1, acc1);
  }
  const float b0v = bo[nt0 * 16 + nn];
  const float b1v = bo[(nt0 + 1) * 16 + nn];
#pragma unroll
  for (int r = 0; r < 8; ++r) {
    int m = hh * 8 + r;
    int grow = row0 + m;
    xs[m * DS_ + nt0 * 16 + nn] =
        acc0[r] + b0v + resq[(size_t)grow * E_ + nt0 * 16 + nn];
    xs[m * DS_ + (nt0 + 1) * 16 + nn] =
        acc1[r] + b1v + resq[(size_t)grow * E_ + (nt0 + 1) * 16 + nn];
  }
  __syncthreads();

  const int row = tid >> 4;
  const int l16 = tid & 15;
  float s = 0.f, ss = 0.f;
#pragma unroll
  for (int i = 0; i < 16; ++i) {
    float x = xs[row * DS_ + l16 * 16 + i];
    s += x; ss += x * x;
  }
  red[tid] = s; red[256 + tid] = ss;
  __syncthreads();
  float S = 0.f, SS = 0.f;
#pragma unroll
  for (int i = 0; i < 16; ++i) { S += red[row * 16 + i]; SS += red[256 + row * 16 + i]; }
  float mu   = S * (1.f / DS_);
  float var  = SS * (1.f / DS_) - mu * mu;
  float rstd = rsqrtf(var + 1e-5f);
#pragma unroll
  for (int i = 0; i < 16; ++i) {
    int col = l16 * 16 + i;
    float x = xs[row * DS_ + col];
    out[(size_t)(row0 + row) * DS_ + col] = (x - mu) * rstd * gamma[col] + beta[col];
  }
}

extern "C" void kernel_launch(void* const* d_in, const int* in_sizes, int n_in,
                              void* d_out, int out_size, void* d_ws, size_t ws_size,
                              hipStream_t stream) {
  const float* q     = (const float*)d_in[0];
  const float* k     = (const float*)d_in[1];
  const float* v     = (const float*)d_in[2];
  const unsigned char* mask = (const unsigned char*)d_in[3];
  const float* Wq    = (const float*)d_in[4];
  const float* bq    = (const float*)d_in[5];
  const float* Wk    = (const float*)d_in[6];
  const float* bk    = (const float*)d_in[7];
  const float* Wv    = (const float*)d_in[8];
  const float* bv    = (const float*)d_in[9];
  const float* Wo    = (const float*)d_in[10];
  const float* bo    = (const float*)d_in[11];
  const float* gamma = (const float*)d_in[12];
  const float* beta  = (const float*)d_in[13];

  float* out      = (float*)d_out;                       // (B,L,DS)
  float* attn_out = out + (size_t)BL_ * DS_;             // (B,H,L,L)

  // workspace carve-out (~78 MB)
  char* ws = (char*)d_ws;
  size_t off = 0;
  auto alloc = [&](size_t bytes) -> char* {
    char* p = ws + off;
    off += (bytes + 255) & ~(size_t)255;
    return p;
  };
  bf16_t* qb  = (bf16_t*)alloc((size_t)BL_ * E_ * 2);
  bf16_t* kb  = (bf16_t*)alloc((size_t)BL_ * E_ * 2);
  bf16_t* vb  = (bf16_t*)alloc((size_t)BL_ * E_ * 2);
  bf16_t* wqT = (bf16_t*)alloc((size_t)EH_ * E_ * 2);
  bf16_t* wkT = (bf16_t*)alloc((size_t)EH_ * E_ * 2);
  bf16_t* wvT = (bf16_t*)alloc((size_t)EH_ * E_ * 2);
  bf16_t* woT = (bf16_t*)alloc((size_t)DS_ * EH_ * 2);
  bf16_t* q_s = (bf16_t*)alloc((size_t)B_ * H_ * L_ * E_ * 2);   // (B,H,L,E)
  bf16_t* k_s = (bf16_t*)alloc((size_t)B_ * H_ * L_ * E_ * 2);   // (B,H,L,E)
  bf16_t* v_t = (bf16_t*)alloc((size_t)B_ * H_ * E_ * L_ * 2);   // (B,H,E,L)
  bf16_t* ctx = (bf16_t*)alloc((size_t)BL_ * EH_ * 2);           // (B,L,H*E)
  (void)ws_size; (void)in_sizes; (void)n_in; (void)out_size;

  // 1) fp32 -> bf16 copies of q/k/v
  {
    int n = BL_ * E_;
    dim3 g(n / 4 / 256);
    cvt_bf16_kernel<<<g, 256, 0, stream>>>(q, qb, n);
    cvt_bf16_kernel<<<g, 256, 0, stream>>>(k, kb, n);
    cvt_bf16_kernel<<<g, 256, 0, stream>>>(v, vb, n);
  }
  // 2) weight transposes to bf16 (K-contiguous B-fragments), tiled/coalesced
  {
    transpose_bf16_kernel<<<dim3(EH_ / 32, E_ / 32), 256, 0, stream>>>(Wq, wqT, E_, EH_);
    transpose_bf16_kernel<<<dim3(EH_ / 32, E_ / 32), 256, 0, stream>>>(Wk, wkT, E_, EH_);
    transpose_bf16_kernel<<<dim3(EH_ / 32, E_ / 32), 256, 0, stream>>>(Wv, wvT, E_, EH_);
    transpose_bf16_kernel<<<dim3(DS_ / 32, EH_ / 32), 256, 0, stream>>>(Wo, woT, EH_, DS_);
  }
  // 3) QKV projections (head-split; V transposed)
  {
    dim3 g(BL_ / 16, EH_ / 128);
    qkv_proj_kernel<<<g, 256, 0, stream>>>(qb, wqT, bq, q_s, 0);
    qkv_proj_kernel<<<g, 256, 0, stream>>>(kb, wkT, bk, k_s, 0);
    qkv_proj_kernel<<<g, 256, 0, stream>>>(vb, wvT, bv, v_t, 1);
  }
  // 4) fused attention (192KB+ dynamic LDS per workgroup — CDNA5 WGP has 320KB)
  {
    hipFuncSetAttribute((const void*)attn_kernel,
                        hipFuncAttributeMaxDynamicSharedMemorySize, SMEM_ATTN);
    dim3 g(L_ / 16, B_ * H_);
    attn_kernel<<<g, 256, SMEM_ATTN, stream>>>(q_s, k_s, v_t, mask, attn_out, ctx);
  }
  // 5) output projection + residual + LayerNorm
  {
    out_proj_ln_kernel<<<dim3(BL_ / 16), 256, 0, stream>>>(ctx, woT, bo, q, gamma,
                                                           beta, out);
  }
}
